// PointNetPlusPlusCls_MSG_5549097746743
// MI455X (gfx1250) — matrix operand order repaired and checked by reading, not compiled, not run on hardware
//
#include <hip/hip_runtime.h>

typedef __attribute__((ext_vector_type(16))) _Float16 v16h;
typedef __attribute__((ext_vector_type(8)))  float    v8f;

// ---------------------------------------------------------------------------
// point_cloud (B,6,N) -> xyz (B,N,3) f32, feat (B,N,3) f32
__global__ void k_split_pc(const float* __restrict__ pc, float* __restrict__ xyz,
                           float* __restrict__ feat, int B, int N) {
  int t = blockIdx.x * blockDim.x + threadIdx.x;
  int total = B * N * 3;
  if (t >= total) return;
  int c = t % 3, n = (t / 3) % N, b = t / (3 * N);
  xyz [(b * N + n) * 3 + c] = pc[(b * 6 + c    ) * (size_t)N + n];
  feat[(b * N + n) * 3 + c] = pc[(b * 6 + 3 + c) * (size_t)N + n];
}

// ---------------------------------------------------------------------------
// Farthest point sampling. One block (256 threads) per batch. N multiple of 256.
__global__ void k_fps(const float* __restrict__ xyz, int* __restrict__ idx,
                      int N, int S) {
  int b = blockIdx.x;
  int tid = threadIdx.x;
  int PPT = N >> 8;
  float dist[16];
  for (int p = 0; p < 16; ++p) dist[p] = 1e10f;
  __shared__ float sv[256];
  __shared__ int   si[256];
  __shared__ int   sfar;
  if (tid == 0) sfar = 0;
  __syncthreads();
  for (int i = 0; i < S; ++i) {
    int far = sfar;
    if (tid == 0) idx[b * S + i] = far;
    float cx = xyz[((size_t)b * N + far) * 3 + 0];
    float cy = xyz[((size_t)b * N + far) * 3 + 1];
    float cz = xyz[((size_t)b * N + far) * 3 + 2];
    float best = -1.0f; int bidx = 0x7fffffff;
    for (int p = 0; p < PPT; ++p) {
      int n = p * 256 + tid;
      float dx = xyz[((size_t)b * N + n) * 3 + 0] - cx;
      float dy = xyz[((size_t)b * N + n) * 3 + 1] - cy;
      float dz = xyz[((size_t)b * N + n) * 3 + 2] - cz;
      float d2 = dx * dx + dy * dy + dz * dz;
      float dm = fminf(dist[p], d2);
      dist[p] = dm;
      if (dm > best || (dm == best && n < bidx)) { best = dm; bidx = n; }
    }
    sv[tid] = best; si[tid] = bidx;
    __syncthreads();
    for (int off = 128; off > 0; off >>= 1) {
      if (tid < off) {
        float v2 = sv[tid + off]; int i2 = si[tid + off];
        if (v2 > sv[tid] || (v2 == sv[tid] && i2 < si[tid])) { sv[tid] = v2; si[tid] = i2; }
      }
      __syncthreads();
    }
    if (tid == 0) sfar = si[0];
    __syncthreads();
  }
}

// ---------------------------------------------------------------------------
__global__ void k_gather_xyz(const float* __restrict__ xyz, const int* __restrict__ idx,
                             float* __restrict__ out, int N, int S, int B) {
  int t = blockIdx.x * blockDim.x + threadIdx.x;
  if (t >= B * S * 3) return;
  int c = t % 3, s = (t / 3) % S, b = t / (3 * S);
  out[(b * S + s) * 3 + c] = xyz[((size_t)b * N + idx[b * S + s]) * 3 + c];
}

// ---------------------------------------------------------------------------
// Ball query: first K points (by index) with dist^2 <= r^2, padded with first.
__global__ void k_ball(const float* __restrict__ xyz, const float* __restrict__ nxyz,
                       int* __restrict__ out, int B, int N, int S, int K, float r2) {
  int t = blockIdx.x * blockDim.x + threadIdx.x;
  if (t >= B * S) return;
  int b = t / S;
  float cx = nxyz[t * 3 + 0], cy = nxyz[t * 3 + 1], cz = nxyz[t * 3 + 2];
  const float* xb = xyz + (size_t)b * N * 3;
  int cnt = 0, first = 0;
  bool have = false;
  for (int n = 0; n < N && cnt < K; ++n) {
    float dx = xb[n * 3 + 0] - cx, dy = xb[n * 3 + 1] - cy, dz = xb[n * 3 + 2] - cz;
    if (dx * dx + dy * dy + dz * dz <= r2) {
      if (!have) { first = n; have = true; }
      out[(size_t)t * K + cnt] = n; ++cnt;
    }
  }
  for (; cnt < K; ++cnt) out[(size_t)t * K + cnt] = first;
}

// ---------------------------------------------------------------------------
// Grouped features G[(b,s,k), c] fp16 with K-padded row stride Cpad (zero fill)
__global__ void k_group(const float* __restrict__ pts, const float* __restrict__ xyz,
                        const float* __restrict__ nxyz, const int* __restrict__ idx,
                        _Float16* __restrict__ G, int B, int N, int S, int K,
                        int Cp, int Cpad) {
  size_t t = (size_t)blockIdx.x * blockDim.x + threadIdx.x;
  size_t total = (size_t)B * S * K * Cpad;
  if (t >= total) return;
  int c = (int)(t % Cpad);
  size_t r = t / Cpad;
  int k = (int)(r % K);
  size_t sb = r / K;
  int s = (int)(sb % S);
  int b = (int)(sb / S);
  float v = 0.0f;
  if (c < Cp + 3) {
    int i = idx[((size_t)b * S + s) * K + k];
    if (c < Cp) v = pts[((size_t)b * N + i) * Cp + c];
    else {
      int d = c - Cp;
      v = xyz[((size_t)b * N + i) * 3 + d] - nxyz[((size_t)b * S + s) * 3 + d];
    }
  }
  G[t] = (_Float16)v;
}

// ---------------------------------------------------------------------------
__global__ void k_wstage(const float* __restrict__ W, _Float16* __restrict__ W16,
                         int Nsrc, int Kd, int Npad, int Kpad) {
  int t = blockIdx.x * blockDim.x + threadIdx.x;
  if (t >= Npad * Kpad) return;
  int k = t % Kpad, n = t / Kpad;
  float v = (n < Nsrc && k < Kd) ? W[(size_t)n * Kd + k] : 0.0f;
  W16[t] = (_Float16)v;
}

__global__ void k_zero_f16(_Float16* __restrict__ d, int n) {
  int t = blockIdx.x * blockDim.x + threadIdx.x;
  if (t < n) d[t] = (_Float16)0.0f;
}

// ---------------------------------------------------------------------------
// WMMA GEMM. X: Mpad x Kpad fp16 (Kpad%32==0, Mpad%16==0). W16: Npad x Kpad fp16
// (Npad%32==0). Block: 8 waves = 8 consecutive M-tiles sharing one 32-column
// B panel staged in LDS. A is double-buffered from global; B frags via ds_load.
// mode 0: y = relu((acc+b)*g/sqrt(1+eps)+beta) -> fp16   mode 2: acc+b+6 -> f32
__global__ __launch_bounds__(256)
void k_gemm(const _Float16* __restrict__ X, const _Float16* __restrict__ W16,
            const float* __restrict__ bias, const float* __restrict__ gamma,
            const float* __restrict__ beta, _Float16* __restrict__ Yh,
            float* __restrict__ Yf, int Mpad, int Kpad, int Npad,
            int Mstore, int Nstore, int ldY, int mode) {
  extern __shared__ _Float16 ldsB[];          // 32 x Kpad panel
  int wave = threadIdx.x >> 5;
  int lane = threadIdx.x & 31;
  int numMT = Mpad >> 4, numNT2 = Npad >> 5;
  int nt  = blockIdx.x % numNT2;
  int mtg = blockIdx.x / numNT2;

  // ---- cooperative B panel copy: global -> LDS (b128 both sides) ----
  {
    const int4* src = (const int4*)(W16 + (size_t)(nt << 5) * Kpad);
    int4* dst = (int4*)ldsB;
    int n4 = (Kpad << 2);                     // 32*Kpad halves / 8 per int4
    for (int i = threadIdx.x; i < n4; i += 256) dst[i] = src[i];
  }
  __syncthreads();

  int mt = (mtg << 3) + wave;
  if (mt >= numMT) return;                    // wave-uniform; after barrier

  int h = lane >> 4, l = lane & 15;
  int row = (mt << 4) + l;
  const _Float16* Xr = X + (size_t)row * Kpad;
  const _Float16* L0 = ldsB + (size_t)l        * Kpad;   // col0 panel row
  const _Float16* L1 = ldsB + (size_t)(l + 16) * Kpad;   // col1 panel row

  union Frag { v16h v; int4 q[2]; } Ac, An, B0, B1;
  union Acc  { v8f  v; float e[8]; } C0, C1;
#pragma unroll
  for (int v = 0; v < 8; ++v) { C0.e[v] = 0.0f; C1.e[v] = 0.0f; }

  // A 16x32 layout: halves j -> K = (j/8)*16 + h*8 + (j%8) => two b128 runs
  Ac.q[0] = *(const int4*)(Xr + (h << 3));
  Ac.q[1] = *(const int4*)(Xr + 16 + (h << 3));

  for (int kb = 0; kb < Kpad; kb += 32) {
    int kn = kb + 32 < Kpad ? kb + 32 : kb;   // branch-free next-k (clamped)
    __builtin_prefetch((const void*)(Xr + kb + 256), 0, 1);
    An.q[0] = *(const int4*)(Xr + kn + (h << 3));
    An.q[1] = *(const int4*)(Xr + kn + 16 + (h << 3));
    // B 32x16 layout: halves j -> K = (j/8)*16 + h*8 + (j%8) from LDS
    const int4* p0 = (const int4*)(L0 + kb + (h << 3));
    B0.q[0] = p0[0]; B0.q[1] = p0[2];
    const int4* p1 = (const int4*)(L1 + kb + (h << 3));
    B1.q[0] = p1[0]; B1.q[1] = p1[2];
    C0.v = __builtin_amdgcn_wmma_f32_16x16x32_f16(false, Ac.v, false, B0.v,
                                                  (short)0, C0.v, false, false);
    C1.v = __builtin_amdgcn_wmma_f32_16x16x32_f16(false, Ac.v, false, B1.v,
                                                  (short)0, C1.v, false, false);
    Ac.q[0] = An.q[0]; Ac.q[1] = An.q[1];
  }

  const float rs = rsqrtf(1.0f + 1e-5f);
  int col0 = (nt << 5) + l;
#pragma unroll
  for (int sub = 0; sub < 2; ++sub) {
    int col = col0 + (sub << 4);
    if (col >= Nstore) continue;
    float bb = bias[col];
    float sc = 0.0f, bt = 0.0f;
    if (mode == 0) { sc = gamma[col] * rs; bt = beta[col]; }
#pragma unroll
    for (int v = 0; v < 8; ++v) {
      int m = (mt << 4) + v + (h << 3);       // f32 C/D layout: M = v + 8*h
      if (m >= Mstore) continue;
      float y = sub ? C1.e[v] : C0.e[v];
      if (mode == 0) {
        y = fmaxf((y + bb) * sc + bt, 0.0f);
        Yh[(size_t)m * ldY + col] = (_Float16)y;
      } else {
        Yf[(size_t)m * ldY + col] = y + bb + 6.0f;
      }
    }
  }
}

// ---------------------------------------------------------------------------
__global__ void k_maxpool(const _Float16* __restrict__ Y, float* __restrict__ out,
                          int GS, int K, int C, int Ctot, int coff) {
  int t = blockIdx.x * blockDim.x + threadIdx.x;
  if (t >= GS * C) return;
  int c = t % C, g = t / C;
  const _Float16* p = Y + (size_t)g * K * C + c;
  float m = -1e30f;
  for (int k = 0; k < K; ++k) m = fmaxf(m, (float)p[(size_t)k * C]);
  out[(size_t)g * Ctot + coff + c] = m;
}

// ---------------------------------------------------------------------------
__global__ void k_cat_xyz_pts(const float* __restrict__ nxyz, const float* __restrict__ pts,
                              _Float16* __restrict__ X, int R, int Cp, int Cpad) {
  size_t t = (size_t)blockIdx.x * blockDim.x + threadIdx.x;
  if (t >= (size_t)R * Cpad) return;
  int c = (int)(t % Cpad);
  int r = (int)(t / Cpad);
  float v = 0.0f;
  if (c < 3) v = nxyz[(size_t)r * 3 + c];
  else if (c < Cp + 3) v = pts[(size_t)r * Cp + (c - 3)];
  X[t] = (_Float16)v;
}

__global__ void k_f32_to_f16(const float* __restrict__ s, _Float16* __restrict__ d, int n) {
  int t = blockIdx.x * blockDim.x + threadIdx.x;
  if (t < n) d[t] = (_Float16)s[t];
}

__global__ void k_copy_f32(const float* __restrict__ s, float* __restrict__ d, int n) {
  int t = blockIdx.x * blockDim.x + threadIdx.x;
  if (t < n) d[t] = s[t];
}

// ---------------------------------------------------------------------------
extern "C" void kernel_launch(void* const* d_in, const int* in_sizes, int n_in,
                              void* d_out, int out_size, void* d_ws, size_t ws_size,
                              hipStream_t stream) {
  (void)in_sizes; (void)n_in; (void)out_size; (void)ws_size;
  const int B = 8, N0 = 4096;
  const float* pc = (const float*)d_in[0];

  struct Layer { const float *W, *b, *g, *be; };
  int pi = 1;
  auto next = [&](Layer& L) {
    L.W  = (const float*)d_in[pi++]; L.b  = (const float*)d_in[pi++];
    L.g  = (const float*)d_in[pi++]; L.be = (const float*)d_in[pi++];
  };
  Layer sa1[3][3], sa2[3][3], sa3L[3], clsL[2];
  for (int s = 0; s < 3; ++s) for (int l = 0; l < 3; ++l) next(sa1[s][l]);
  for (int s = 0; s < 3; ++s) for (int l = 0; l < 3; ++l) next(sa2[s][l]);
  for (int l = 0; l < 3; ++l) next(sa3L[l]);
  for (int l = 0; l < 2; ++l) next(clsL[l]);
  const float* Wf = (const float*)d_in[pi++];
  const float* bf = (const float*)d_in[pi++];

  // ---- workspace carve-out ----
  char* base = (char*)d_ws; size_t off = 0;
  auto carve = [&](size_t bytes) -> void* {
    void* p = base + off; off = (off + bytes + 255) & ~(size_t)255; return p;
  };
  float*    xyz    = (float*)carve((size_t)B * N0 * 3 * 4);
  float*    feat   = (float*)carve((size_t)B * N0 * 3 * 4);
  int*      fpsIdx = (int*)  carve((size_t)B * 512 * 4);
  float*    nx1    = (float*)carve((size_t)B * 512 * 3 * 4);
  float*    nx2    = (float*)carve((size_t)B * 128 * 3 * 4);
  int*      ballIdx= (int*)  carve((size_t)B * 512 * 128 * 4);
  float*    pts1   = (float*)carve((size_t)B * 512 * 320 * 4);
  float*    pts2   = (float*)carve((size_t)B * 128 * 640 * 4);
  float*    feat3  = (float*)carve((size_t)B * 1024 * 4);
  _Float16* xh0    = (_Float16*)carve((size_t)16 * 1024 * 2);
  _Float16* xh1    = (_Float16*)carve((size_t)16 * 1024 * 2);
  const size_t BIG = (size_t)524288 * 128;              // halves
  _Float16* bufA   = (_Float16*)carve(BIG * 2);
  _Float16* bufB   = (_Float16*)carve(BIG * 2);

  dim3 blk(256);
  auto nb = [](size_t n) { return dim3((unsigned)((n + 255) / 256)); };

  auto stageW = [&](const float* W, int Nsrc, int Kd, int Npad, int Kpad) -> _Float16* {
    _Float16* w16 = (_Float16*)carve((size_t)Npad * Kpad * 2);
    k_wstage<<<nb((size_t)Npad * Kpad), blk, 0, stream>>>(W, w16, Nsrc, Kd, Npad, Kpad);
    return w16;
  };
  auto gemm = [&](const _Float16* X, const _Float16* W16, const float* b,
                  const float* g, const float* be, _Float16* Yh, float* Yf,
                  int Mpad, int Kpad, int Npad, int Mstore, int Nstore, int ldY, int mode) {
    int numMT = Mpad >> 4, numNT2 = Npad >> 5;
    int mtBlocks = (numMT + 7) >> 3;
    unsigned grid = (unsigned)mtBlocks * (unsigned)numNT2;
    size_t lds = (size_t)Kpad * 64;           // 32 cols * Kpad halves * 2B
    k_gemm<<<dim3(grid), dim3(256), lds, stream>>>(
        X, W16, b, g, be, Yh, Yf, Mpad, Kpad, Npad, Mstore, Nstore, ldY, mode);
  };

  k_split_pc<<<nb((size_t)B * N0 * 3), blk, 0, stream>>>(pc, xyz, feat, B, N0);

  // ---------------- SA1 (MSG), Cin 6 -> pad 32 ----------------
  k_fps<<<B, 256, 0, stream>>>(xyz, fpsIdx, N0, 512);
  k_gather_xyz<<<nb((size_t)B * 512 * 3), blk, 0, stream>>>(xyz, fpsIdx, nx1, N0, 512, B);
  {
    const int   Ks[3] = {16, 32, 128};
    const float rs[3] = {0.1f, 0.2f, 0.4f};
    const int   co[3][3] = {{32, 32, 64}, {64, 64, 128}, {64, 96, 128}};
    int coff = 0;
    for (int sc = 0; sc < 3; ++sc) {
      int K = Ks[sc], S = 512, Cpad = 32;
      int M = B * S * K;
      k_ball<<<nb((size_t)B * S), blk, 0, stream>>>(xyz, nx1, ballIdx, B, N0, S, K, rs[sc] * rs[sc]);
      k_group<<<nb((size_t)M * Cpad), blk, 0, stream>>>(feat, xyz, nx1, ballIdx, bufA,
                                                        B, N0, S, K, 3, Cpad);
      const _Float16* cur = bufA; int cin = Cpad, kd = 6;
      for (int l = 0; l < 3; ++l) {
        int cout = co[sc][l];
        _Float16* nxt = (cur == bufA) ? bufB : bufA;
        _Float16* w16 = stageW(sa1[sc][l].W, cout, kd, cout, cin);
        gemm(cur, w16, sa1[sc][l].b, sa1[sc][l].g, sa1[sc][l].be, nxt, nullptr,
             M, cin, cout, M, cout, cout, 0);
        cur = nxt; cin = cout; kd = cout;
      }
      k_maxpool<<<nb((size_t)B * S * cin), blk, 0, stream>>>(cur, pts1, B * S, K, cin, 320, coff);
      coff += cin;
    }
  }

  // ---------------- SA2 (MSG), Cin 323 -> pad 352 ----------------
  k_fps<<<B, 256, 0, stream>>>(nx1, fpsIdx, 512, 128);
  k_gather_xyz<<<nb((size_t)B * 128 * 3), blk, 0, stream>>>(nx1, fpsIdx, nx2, 512, 128, B);
  {
    const int   Ks[3] = {32, 64, 128};
    const float rs[3] = {0.2f, 0.4f, 0.8f};
    const int   co[3][3] = {{64, 64, 128}, {128, 128, 256}, {128, 128, 256}};
    int coff = 0;
    for (int sc = 0; sc < 3; ++sc) {
      int K = Ks[sc], S = 128, Cpad = 352, Nsrc = 512;
      int M = B * S * K;
      k_ball<<<nb((size_t)B * S), blk, 0, stream>>>(nx1, nx2, ballIdx, B, Nsrc, S, K, rs[sc] * rs[sc]);
      k_group<<<nb((size_t)M * Cpad), blk, 0, stream>>>(pts1, nx1, nx2, ballIdx, bufA,
                                                        B, Nsrc, S, K, 320, Cpad);
      const _Float16* cur = bufA; int cin = Cpad, kd = 323;
      for (int l = 0; l < 3; ++l) {
        int cout = co[sc][l];
        _Float16* nxt = (cur == bufA) ? bufB : bufA;
        _Float16* w16 = stageW(sa2[sc][l].W, cout, kd, cout, cin);
        gemm(cur, w16, sa2[sc][l].b, sa2[sc][l].g, sa2[sc][l].be, nxt, nullptr,
             M, cin, cout, M, cout, cout, 0);
        cur = nxt; cin = cout; kd = cout;
      }
      k_maxpool<<<nb((size_t)B * S * cin), blk, 0, stream>>>(cur, pts2, B * S, K, cin, 640, coff);
      coff += cin;
    }
  }

  // ---------------- SA3 (group-all), Cin 643 -> pad 672 ----------------
  {
    int R = B * 128, Cpad = 672;
    const int co[3] = {256, 512, 1024};
    k_cat_xyz_pts<<<nb((size_t)R * Cpad), blk, 0, stream>>>(nx2, pts2, bufA, R, 640, Cpad);
    const _Float16* cur = bufA; int cin = Cpad, kd = 643;
    for (int l = 0; l < 3; ++l) {
      int cout = co[l];
      _Float16* nxt = (cur == bufA) ? bufB : bufA;
      _Float16* w16 = stageW(sa3L[l].W, cout, kd, cout, cin);
      gemm(cur, w16, sa3L[l].b, sa3L[l].g, sa3L[l].be, nxt, nullptr,
           R, cin, cout, R, cout, cout, 0);
      cur = nxt; cin = cout; kd = cout;
    }
    k_maxpool<<<nb((size_t)B * 1024), blk, 0, stream>>>(cur, feat3, B, 128, 1024, 1024, 0);
  }

  // ---------------- Classifier (M padded 8 -> 16, zero rows) ----------------
  k_f32_to_f16<<<nb((size_t)B * 1024), blk, 0, stream>>>(feat3, xh0, B * 1024);
  k_zero_f16<<<nb((size_t)8 * 1024), blk, 0, stream>>>(xh0 + (size_t)B * 1024, 8 * 1024);
  {
    _Float16* w0 = stageW(clsL[0].W, 512, 1024, 512, 1024);
    gemm(xh0, w0, clsL[0].b, clsL[0].g, clsL[0].be, xh1, nullptr,
         16, 1024, 512, 16, 512, 512, 0);
    _Float16* w1 = stageW(clsL[1].W, 256, 512, 256, 512);
    gemm(xh1, w1, clsL[1].b, clsL[1].g, clsL[1].be, xh0, nullptr,
         16, 512, 256, 16, 256, 256, 0);
    _Float16* wf = stageW(Wf, 40, 256, 64, 256);
    gemm(xh0, wf, bf, nullptr, nullptr, nullptr, (float*)d_out,
         16, 256, 64, B, 40, 40, 2);
  }

  // points_sa_3 (B,1024,1) flat == feat3 flat
  k_copy_f32<<<nb((size_t)B * 1024), blk, 0, stream>>>(feat3, (float*)d_out + B * 40, B * 1024);
}
